// MultiheadAttention_5961414607140
// MI455X (gfx1250) — matrix operand order repaired and checked by reading, not compiled, not run on hardware
//
#include <hip/hip_runtime.h>

// MultiheadAttention forward for MI455X (gfx1250), bf16 WMMA + fp32 accum.
// B=2, S=2048, D=1024, H=16, HD=64.

#define BB  2
#define SS  2048
#define DD  1024
#define HH  16
#define HDQ 64

typedef __attribute__((ext_vector_type(16))) __bf16 v16bf;
typedef __attribute__((ext_vector_type(8)))  __bf16 v8bf;
typedef __attribute__((ext_vector_type(8)))  float  v8f;

// Native convert (lowers to hardware bf16 cvt on gfx1250, RNE).
static __device__ __forceinline__ __bf16 f2bf(float f) { return (__bf16)f; }

// ---- WMMA fragment loaders (layouts per CDNA5 ISA §7.12.2) ----
// A (16x32, bf16): lane<16 -> M=lane, K = k0+{0..7} then k0+16+{0..7}
//                  lane>=16 -> M=lane-16, K = k0+8+{0..7} then k0+24+{0..7}
static __device__ __forceinline__ v16bf load_a_bf(const __bf16* __restrict__ base,
                                                  int ld, int row0, int k0, int lane) {
  int m = lane & 15, half = lane >> 4;
  const __bf16* p = base + (size_t)(row0 + m) * ld + k0 + half * 8;
  v8bf lo = *(const v8bf*)p;
  v8bf hi = *(const v8bf*)(p + 16);
  return __builtin_shufflevector(lo, hi, 0,1,2,3,4,5,6,7,8,9,10,11,12,13,14,15);
}

// B (32x16, bf16), source stored as [N rows, K cols] row-major (i.e. Bt):
// lane -> N = lane&15 ; K = k0 + (lane>>4)*8 + {0..7} then k0 + 16 + (lane>>4)*8 + {0..7}
static __device__ __forceinline__ v16bf load_b_bf(const __bf16* __restrict__ base,
                                                  int ld, int n0, int k0, int lane) {
  int n = lane & 15, half = lane >> 4;
  const __bf16* p = base + (size_t)(n0 + n) * ld + k0 + half * 8;
  v8bf lo = *(const v8bf*)p;
  v8bf hi = *(const v8bf*)(p + 16);
  return __builtin_shufflevector(lo, hi, 0,1,2,3,4,5,6,7,8,9,10,11,12,13,14,15);
}

static __device__ __forceinline__ v8f wmma_bf16(v16bf a, v16bf b, v8f c) {
  return __builtin_amdgcn_wmma_f32_16x16x32_bf16(false, a, false, b, (short)0, c, false, false);
}

// -------------------- fp32 -> bf16 conversion (8 elems/thread) ----------
__global__ __launch_bounds__(256)
void cvt_f32_bf16(const float* __restrict__ in, __bf16* __restrict__ out, int n) {
  int i = (blockIdx.x * blockDim.x + threadIdx.x) * 8;
  if (i + 7 < n) {
    v8f v = *(const v8f*)(in + i);
    v8bf o;
#pragma unroll
    for (int j = 0; j < 8; ++j) o[j] = f2bf(v[j]);
    *(v8bf*)(out + i) = o;
  }
}

// -------------------- QKV projection:  Y = X @ W^T ----------------------
// X bf16 [B*S, D], W bf16 [D(out), D(in)].  Output bf16:
//   vtrans==0 -> [bh, S, HD] (Q, K);  vtrans==1 -> [bh, HD, S] (V transposed).
// Each wave computes a 16x128 tile (8 WMMA accumulators); block = 4 waves.
__global__ __launch_bounds__(128)
void proj_qkv_kernel(const __bf16* __restrict__ X, const __bf16* __restrict__ W,
                     __bf16* __restrict__ Out, float scale, int vtrans) {
  int lane = threadIdx.x & 31, wave = threadIdx.x >> 5;
  int m0 = blockIdx.x * 64 + wave * 16;
  int n0 = blockIdx.y * 128;
  v8f acc[8] = {};
  for (int k0 = 0; k0 < DD; k0 += 32) {
    v16bf a = load_a_bf(X, DD, m0, k0, lane);
#pragma unroll
    for (int t = 0; t < 8; ++t) {
      v16bf b = load_b_bf(W, DD, n0 + t * 16, k0, lane);
      acc[t] = wmma_bf16(a, b, acc[t]);
    }
  }
  int nl = lane & 15, half = lane >> 4;
#pragma unroll
  for (int t = 0; t < 8; ++t) {
    int n = n0 + t * 16 + nl;
    int h = n >> 6, hd = n & (HDQ - 1);
#pragma unroll
    for (int r = 0; r < 8; ++r) {
      int m = m0 + half * 8 + r;
      int b = m >> 11, s = m & (SS - 1);
      int bh = b * HH + h;
      float v = acc[t][r] * scale;
      if (vtrans) Out[((size_t)bh * HDQ + hd) * SS + s] = f2bf(v);
      else        Out[((size_t)bh * SS + s) * HDQ + hd] = f2bf(v);
    }
  }
}

// -------------------- flash attention (one wave per 16-query tile) ------
// Q,K: [bh, S, 64] bf16 (Q pre-scaled by 1/HD).  Vt: [bh, 64, S] bf16.
// O: [B, S, D] bf16 token-major for the output projection.
__global__ __launch_bounds__(128)
void flash_attn_kernel(const __bf16* __restrict__ Q, const __bf16* __restrict__ K,
                       const __bf16* __restrict__ Vt,
                       const unsigned char* __restrict__ kpm,
                       __bf16* __restrict__ O) {
  __shared__ __bf16 lds_p[4][16][40];   // per-wave 16x32 P staging (+pad, 16B-aligned rows)
  int lane = threadIdx.x & 31, wave = threadIdx.x >> 5;
  int bh = blockIdx.y;
  int b = bh >> 4, h = bh & (HH - 1);
  int q0 = (blockIdx.x * 4 + wave) * 16;
  const __bf16* Qh = Q + (size_t)bh * SS * HDQ;
  const __bf16* Kh = K + (size_t)bh * SS * HDQ;
  const __bf16* Vh = Vt + (size_t)bh * HDQ * SS;
  const unsigned char* mh = kpm + (size_t)b * SS;

  // Q tile (16x64) held in registers: two A fragments.
  v16bf qa0 = load_a_bf(Qh, HDQ, q0, 0, lane);
  v16bf qa1 = load_a_bf(Qh, HDQ, q0, 32, lane);

  float mrow[8], lrow[8];
#pragma unroll
  for (int r = 0; r < 8; ++r) { mrow[r] = -3.0e38f; lrow[r] = 0.f; }
  v8f acc[4] = {};                       // O tile 16x64 fp32

  int nl = lane & 15, half = lane >> 4;
  __bf16 (*lp)[40] = lds_p[wave];

  for (int j0 = 0; j0 < SS; j0 += 32) {
    // scores: two 16x16 tiles over keys [j0, j0+32)
    v8f s0 = {}, s1 = {};
    {
      v16bf kb;
      kb = load_b_bf(Kh, HDQ, j0,      0,  lane); s0 = wmma_bf16(qa0, kb, s0);
      kb = load_b_bf(Kh, HDQ, j0,      32, lane); s0 = wmma_bf16(qa1, kb, s0);
      kb = load_b_bf(Kh, HDQ, j0 + 16, 0,  lane); s1 = wmma_bf16(qa0, kb, s1);
      kb = load_b_bf(Kh, HDQ, j0 + 16, 32, lane); s1 = wmma_bf16(qa1, kb, s1);
    }
    // key-padding mask: this lane owns one key column per tile
    if (!mh[j0 + nl]) {
#pragma unroll
      for (int r = 0; r < 8; ++r) s0[r] = -3.0e38f;
    }
    if (!mh[j0 + 16 + nl]) {
#pragma unroll
      for (int r = 0; r < 8; ++r) s1[r] = -3.0e38f;
    }
    // online softmax: row stats live per 16-lane half (D layout: lane=N, vgpr=M)
    float tmax[8];
#pragma unroll
    for (int r = 0; r < 8; ++r) tmax[r] = fmaxf(s0[r], s1[r]);
#pragma unroll
    for (int off = 1; off < 16; off <<= 1)
#pragma unroll
      for (int r = 0; r < 8; ++r) tmax[r] = fmaxf(tmax[r], __shfl_xor(tmax[r], off));

    float alpha[8], p0[8], p1[8], tsum[8];
#pragma unroll
    for (int r = 0; r < 8; ++r) {
      float mn = fmaxf(mrow[r], tmax[r]);
      alpha[r] = __expf(mrow[r] - mn);
      mrow[r]  = mn;
      p0[r] = __expf(s0[r] - mn);
      p1[r] = __expf(s1[r] - mn);
      tsum[r] = p0[r] + p1[r];
    }
#pragma unroll
    for (int off = 1; off < 16; off <<= 1)
#pragma unroll
      for (int r = 0; r < 8; ++r) tsum[r] += __shfl_xor(tsum[r], off);
#pragma unroll
    for (int r = 0; r < 8; ++r) lrow[r] = lrow[r] * alpha[r] + tsum[r];
#pragma unroll
    for (int t = 0; t < 4; ++t)
#pragma unroll
      for (int r = 0; r < 8; ++r) acc[t][r] *= alpha[r];

    // transpose P (D layout -> A layout) through this wave's LDS tile.
    // LDS ops are in-order within a wave: no barrier needed.
#pragma unroll
    for (int r = 0; r < 8; ++r) {
      lp[half * 8 + r][nl]      = f2bf(p0[r]);
      lp[half * 8 + r][16 + nl] = f2bf(p1[r]);
    }
    const __bf16* pr = &lp[nl][half * 8];
    v8bf plo = *(const v8bf*)pr;
    v8bf phi = *(const v8bf*)(pr + 16);
    v16bf pa = __builtin_shufflevector(plo, phi, 0,1,2,3,4,5,6,7,8,9,10,11,12,13,14,15);

    // acc += P (16x32) x V (32x64); Vt rows are head dims -> contiguous B frags
#pragma unroll
    for (int t = 0; t < 4; ++t) {
      v16bf vb = load_b_bf(Vh, SS, t * 16, j0, lane);
      acc[t] = wmma_bf16(pa, vb, acc[t]);
    }
  }

  // epilogue: divide by row sums, store token-major bf16
  float invl[8];
#pragma unroll
  for (int r = 0; r < 8; ++r) invl[r] = 1.0f / lrow[r];
#pragma unroll
  for (int t = 0; t < 4; ++t) {
#pragma unroll
    for (int r = 0; r < 8; ++r) {
      int s = q0 + half * 8 + r;
      int col = h * HDQ + t * 16 + nl;
      O[((size_t)b * SS + s) * DD + col] = f2bf(acc[t][r] * invl[r]);
    }
  }
}

// -------------------- output projection: Y = O @ Wout^T + b -------------
__global__ __launch_bounds__(128)
void proj_out_kernel(const __bf16* __restrict__ A, const __bf16* __restrict__ W,
                     const float* __restrict__ bias, float* __restrict__ Y) {
  int lane = threadIdx.x & 31, wave = threadIdx.x >> 5;
  int m0 = blockIdx.x * 64 + wave * 16;
  int n0 = blockIdx.y * 128;
  v8f acc[8] = {};
  for (int k0 = 0; k0 < DD; k0 += 32) {
    v16bf a = load_a_bf(A, DD, m0, k0, lane);
#pragma unroll
    for (int t = 0; t < 8; ++t) {
      v16bf b = load_b_bf(W, DD, n0 + t * 16, k0, lane);
      acc[t] = wmma_bf16(a, b, acc[t]);
    }
  }
  int nl = lane & 15, half = lane >> 4;
#pragma unroll
  for (int t = 0; t < 8; ++t) {
    int n = n0 + t * 16 + nl;
    float bv = bias[n];
#pragma unroll
    for (int r = 0; r < 8; ++r) {
      int m = m0 + half * 8 + r;
      Y[(size_t)m * DD + n] = acc[t][r] + bv;
    }
  }
}

// ------------------------------ launcher --------------------------------
extern "C" void kernel_launch(void* const* d_in, const int* in_sizes, int n_in,
                              void* d_out, int out_size, void* d_ws, size_t ws_size,
                              hipStream_t stream) {
  const float* query = (const float*)d_in[0];
  const float* key   = (const float*)d_in[1];
  const float* value = (const float*)d_in[2];
  const unsigned char* kpm = (const unsigned char*)d_in[3]; // bool mask [B,S]
  const float* wq    = (const float*)d_in[4];
  const float* wk    = (const float*)d_in[5];
  const float* wv    = (const float*)d_in[6];
  const float* w_out = (const float*)d_in[7];
  const float* b_out = (const float*)d_in[8];
  float* out = (float*)d_out;

  // scratch layout (bf16 elements):
  //   4 weights (1M each) + Xq,Xk,Xv (4M each) + Q,K,Vt,Ob (4M each) = 32M bf16 = 64 MB
  __bf16* ws = (__bf16*)d_ws;
  const size_t WSZ = (size_t)DD * DD;
  const size_t TSZ = (size_t)BB * SS * DD;
  __bf16* wq_bf = ws;
  __bf16* wk_bf = wq_bf + WSZ;
  __bf16* wv_bf = wk_bf + WSZ;
  __bf16* wo_bf = wv_bf + WSZ;
  __bf16* Xq = wo_bf + WSZ;   // activations, bf16
  __bf16* Xk = Xq + TSZ;
  __bf16* Xv = Xk + TSZ;
  __bf16* Qp = Xv + TSZ;      // [bh, S, HD], pre-scaled by 1/HD
  __bf16* Kp = Qp + TSZ;      // [bh, S, HD]
  __bf16* Vt = Kp + TSZ;      // [bh, HD, S]
  __bf16* Ob = Vt + TSZ;      // [B, S, D]

  {
    int nw = DD * DD;                       // 1M elems per weight
    dim3 gw((unsigned)(nw / 8 / 256));
    cvt_f32_bf16<<<gw, 256, 0, stream>>>(wq, wq_bf, nw);
    cvt_f32_bf16<<<gw, 256, 0, stream>>>(wk, wk_bf, nw);
    cvt_f32_bf16<<<gw, 256, 0, stream>>>(wv, wv_bf, nw);
    cvt_f32_bf16<<<gw, 256, 0, stream>>>(w_out, wo_bf, nw);
    int nx = BB * SS * DD;                  // 4M elems per activation tensor
    dim3 gx((unsigned)(nx / 8 / 256));
    cvt_f32_bf16<<<gx, 256, 0, stream>>>(query, Xq, nx);
    cvt_f32_bf16<<<gx, 256, 0, stream>>>(key,   Xk, nx);
    cvt_f32_bf16<<<gx, 256, 0, stream>>>(value, Xv, nx);
  }

  dim3 gp(BB * SS / 64, DD / 128);  // (64, 8), 128 threads = 4 waves
  // reference scales q by 1/sqrt(HD) and scores by HD^-0.5 -> total 1/HD
  proj_qkv_kernel<<<gp, 128, 0, stream>>>(Xq, wq_bf, Qp, 1.0f / (float)HDQ, 0);
  proj_qkv_kernel<<<gp, 128, 0, stream>>>(Xk, wk_bf, Kp, 1.0f, 0);
  proj_qkv_kernel<<<gp, 128, 0, stream>>>(Xv, wv_bf, Vt, 1.0f, 1);

  dim3 gf(SS / 64, BB * HH);        // (32, 32): 4 q-tiles per block x 32 bh
  flash_attn_kernel<<<gf, 128, 0, stream>>>(Qp, Kp, Vt, kpm, Ob);

  proj_out_kernel<<<gp, 128, 0, stream>>>(Ob, wo_bf, b_out, out);
}
